// GIN_Model_12549894439566
// MI455X (gfx1250) — matrix operand order repaired and checked by reading, not compiled, not run on hardware
//
#include <hip/hip_runtime.h>
#include <hip/hip_bf16.h>
#include <stdint.h>

// Fused DenseGraphFlow + GraphAttention + LayerNorm ensemble for MI455X (gfx1250).
// One workgroup (8 wave32) per batch element; all intermediates live in LDS;
// all GEMMs run on v_wmma_f32_16x16x32_bf16 with fp32 accumulation.

#define NND   64     // nodes
#define FINN  96
#define FOUTN 128
#define OPDP  64     // OPD=48 zero-padded to 64 so K is a multiple of 32
#define EPSL  1e-5f
#define SLOPE 0.01f

typedef __attribute__((ext_vector_type(16))) __bf16       bf16x16;
typedef __attribute__((ext_vector_type(8)))  float        f32x8;
typedef __attribute__((ext_vector_type(8)))  unsigned int u32x8;

union FragU { u32x8 u; bf16x16 b; };

struct Smem {
  unsigned short H[NND * FINN];        // h[b]      [i][f]
  unsigned short Adj[NND * NND];       // adj[b]    [i][j]
  unsigned short Op[NND * OPDP];       // op_emb[b] [i][c], c>=48 zero
  unsigned short Wt[FOUTN * FINN];     // W^T   [o][f]
  unsigned short Wkt[FOUTN * FINN];    // Wk^T  [o][f]
  unsigned short WopDt[FOUTN * OPDP];  // Wop_d^T [o][c] padded
  unsigned short WopGt[FOUTN * OPDP];  // Wop_g^T [o][c] padded
  unsigned short SupT[FOUTN * NND];    // support^T [o][i]
  unsigned short WhkAW[NND * FOUTN];   // (Whk*a_w) [i][m]  (A of scores)
  unsigned short Whk[NND * FOUTN];     // Whk [j][m]        (B^T of scores)
  unsigned short WhkT[FOUTN * NND];    // Whk^T [o][j]      (B^T of hp GEMM)
  unsigned short Attn[NND * NND];      // softmax result, bf16
  float ScoresF[NND * NND];
  float HpF[NND * FOUTN];
  float mu[NND];
  float rstd[NND];
  float bvec[FOUTN], awv[FOUTN], gam[FOUTN], bet[FOUTN], bopd[FOUTN], bopg[FOUTN];
};

__device__ __forceinline__ unsigned short f2bf(float f) {
  union { float f; unsigned u; } c; c.f = f;
  unsigned u = c.u;
  u += 0x7FFFu + ((u >> 16) & 1u);        // round-to-nearest-even
  return (unsigned short)(u >> 16);
}
__device__ __forceinline__ float bf2f(unsigned short b) {
  union { unsigned u; float f; } c; c.u = ((unsigned)b) << 16;
  return c.f;
}
__device__ __forceinline__ f32x8 zero8() {
  f32x8 z;
#pragma unroll
  for (int e = 0; e < 8; ++e) z[e] = 0.f;
  return z;
}

// A fragment: 16x32 bf16 tile at (mBase,kBase) of row-major [rows][ld] matrix.
// ISA 7.12.2: lane(hi,l) row M=l; VGPR v<4 -> K=2v+8hi, v>=4 -> K=16+2(v-4)+8hi.
__device__ __forceinline__ bf16x16 load_frag_a(const unsigned short* src, int ld,
                                               int mBase, int kBase, int l, int hi) {
  FragU f;
  const unsigned int* p = (const unsigned int*)(src + (mBase + l) * ld + kBase + 8 * hi);
  f.u[0] = p[0]; f.u[1] = p[1]; f.u[2] = p[2];  f.u[3] = p[3];
  f.u[4] = p[8]; f.u[5] = p[9]; f.u[6] = p[10]; f.u[7] = p[11];
  return f.b;
}

// B fragment: 32x16 tile (KxN) at (kBase,nBase); operand stored transposed [N][ld]
// so K runs contiguously. ISA: lane(hi,n) col N=n; VGPR v<4 -> K=2v+8hi, v>=4 -> K=16+2(v-4)+8hi.
__device__ __forceinline__ bf16x16 load_frag_b(const unsigned short* srcT, int ld,
                                               int kBase, int nBase, int l, int hi) {
  FragU f;
  const unsigned int* p = (const unsigned int*)(srcT + (nBase + l) * ld + kBase + 8 * hi);
  f.u[0] = p[0]; f.u[1] = p[1]; f.u[2] = p[2];  f.u[3] = p[3];
  f.u[4] = p[8]; f.u[5] = p[9]; f.u[6] = p[10]; f.u[7] = p[11];
  return f.b;
}

__device__ __forceinline__ f32x8 wmma_bf16(bf16x16 a, bf16x16 b, f32x8 c) {
  return __builtin_amdgcn_wmma_f32_16x16x32_bf16(false, a, false, b, (short)0, c,
                                                 false, false);
}

__global__ void __launch_bounds__(256, 1)
gin_fused_kernel(const float* __restrict__ hG, const float* __restrict__ adjG,
                 const float* __restrict__ opG,
                 const float* __restrict__ Wg,  const float* __restrict__ bG,
                 const float* __restrict__ WopDg, const float* __restrict__ bopDg,
                 const float* __restrict__ Wkg, const float* __restrict__ awG,
                 const float* __restrict__ WopGg, const float* __restrict__ bopGg,
                 const float* __restrict__ gamG, const float* __restrict__ betG,
                 float* __restrict__ outG) {
  extern __shared__ char smem_raw[];
  Smem& s = *reinterpret_cast<Smem*>(smem_raw);

  const int tid  = threadIdx.x;
  const int bId  = blockIdx.x;
  const int wv   = tid >> 5;
  const int lane = tid & 31;
  const int l    = lane & 15;
  const int hi   = lane >> 4;

  const float* hB   = hG   + (size_t)bId * NND * FINN;
  const float* adjB = adjG + (size_t)bId * NND * NND;
  const float* opB  = opG  + (size_t)bId * NND * 48;

  // ---------------- Phase 0: stage everything to LDS as bf16 ----------------
  for (int i = tid; i < NND * FINN; i += 256) s.H[i]   = f2bf(hB[i]);
  for (int i = tid; i < NND * NND;  i += 256) s.Adj[i] = f2bf(adjB[i]);
  for (int i = tid; i < NND * OPDP; i += 256) {
    int r = i >> 6, c = i & 63;
    s.Op[i] = (c < 48) ? f2bf(opB[r * 48 + c]) : (unsigned short)0;
  }
  for (int i = tid; i < FINN * FOUTN; i += 256) {       // W, Wk are [f][o]
    int f = i >> 7, o = i & 127;
    s.Wt[o * FINN + f]  = f2bf(Wg[i]);
    s.Wkt[o * FINN + f] = f2bf(Wkg[i]);
  }
  for (int i = tid; i < 48 * FOUTN; i += 256) {          // Wop_* are [c][o]
    int c = i >> 7, o = i & 127;
    s.WopDt[o * OPDP + c] = f2bf(WopDg[i]);
    s.WopGt[o * OPDP + c] = f2bf(WopGg[i]);
  }
  for (int i = tid; i < FOUTN * 16; i += 256) {          // zero-fill pad c=48..63
    int o = i >> 4, c = 48 + (i & 15);
    s.WopDt[o * OPDP + c] = 0;
    s.WopGt[o * OPDP + c] = 0;
  }
  if (tid < FOUTN) {
    s.bvec[tid] = bG[tid];    s.awv[tid]  = awG[tid];
    s.gam[tid]  = gamG[tid];  s.bet[tid]  = betG[tid];
    s.bopd[tid] = bopDg[tid]; s.bopg[tid] = bopGg[tid];
  }
  __syncthreads();

  // Tile ownership: wave -> M block (w&3), N blocks (w>>2)*4 .. +3 (for 64x128 GEMMs)
  const int mBase = (wv & 3) * 16;
  const int nb0   = (wv >> 2) * 4;

  // -------- Phase A: support = h@W and Whk = h@Wk (K = 96 = 3x32) --------
  f32x8 accS[4], accK[4];
#pragma unroll
  for (int t = 0; t < 4; ++t) { accS[t] = zero8(); accK[t] = zero8(); }
#pragma unroll
  for (int ks = 0; ks < 3; ++ks) {
    bf16x16 a = load_frag_a(s.H, FINN, mBase, ks * 32, l, hi);
#pragma unroll
    for (int t = 0; t < 4; ++t) {
      bf16x16 bS = load_frag_b(s.Wt,  FINN, ks * 32, (nb0 + t) * 16, l, hi);
      accS[t] = wmma_bf16(a, bS, accS[t]);
      bf16x16 bK = load_frag_b(s.Wkt, FINN, ks * 32, (nb0 + t) * 16, l, hi);
      accK[t] = wmma_bf16(a, bK, accK[t]);
    }
  }
  // Emit bf16 derived operands (support^T, Whk, Whk*a_w, Whk^T). Keep accS (fp32).
#pragma unroll
  for (int t = 0; t < 4; ++t) {
    const int o = (nb0 + t) * 16 + l;
#pragma unroll
    for (int r = 0; r < 8; ++r) {
      const int i = mBase + r + 8 * hi;
      const float kv = accK[t][r];
      s.SupT[o * NND + i]      = f2bf(accS[t][r]);
      s.WhkAW[i * FOUTN + o]   = f2bf(kv * s.awv[o]);
      s.Whk[i * FOUTN + o]     = f2bf(kv);
      s.WhkT[o * NND + i]      = f2bf(kv);
    }
  }
  __syncthreads();

  // -------- Phase B: adjS = adj @ support (K = 64 = 2x32), kept in regs --------
  f32x8 accA[4];
#pragma unroll
  for (int t = 0; t < 4; ++t) accA[t] = zero8();
#pragma unroll
  for (int ks = 0; ks < 2; ++ks) {
    bf16x16 a = load_frag_a(s.Adj, NND, mBase, ks * 32, l, hi);
#pragma unroll
    for (int t = 0; t < 4; ++t) {
      bf16x16 bS = load_frag_b(s.SupT, NND, ks * 32, (nb0 + t) * 16, l, hi);
      accA[t] = wmma_bf16(a, bS, accA[t]);
    }
  }

  // -------- Phase C: scores = (Whk*a_w) @ Whk^T (64x64, K = 128 = 4x32) --------
  {
    const int nbs = (wv >> 2) * 2;   // 2 tiles/wave over 4 N-blocks
    f32x8 accSc[2];
    accSc[0] = zero8(); accSc[1] = zero8();
#pragma unroll
    for (int ks = 0; ks < 4; ++ks) {
      bf16x16 a = load_frag_a(s.WhkAW, FOUTN, mBase, ks * 32, l, hi);
#pragma unroll
      for (int t2 = 0; t2 < 2; ++t2) {
        bf16x16 bb = load_frag_b(s.Whk, FOUTN, ks * 32, (nbs + t2) * 16, l, hi);
        accSc[t2] = wmma_bf16(a, bb, accSc[t2]);
      }
    }
#pragma unroll
    for (int t2 = 0; t2 < 2; ++t2) {
      const int j = (nbs + t2) * 16 + l;
#pragma unroll
      for (int r = 0; r < 8; ++r) {
        const int i = mBase + r + 8 * hi;
        s.ScoresF[i * NND + j] = accSc[t2][r];
      }
    }
  }
  __syncthreads();

  // -------- Phase D: alpha = leaky_relu(scores)*adj; softmax over ALL j --------
  {
    const int row = tid >> 2;
    const int c0  = (tid & 3) * 16;
    float av[16];
    float m = -3.4e38f;
#pragma unroll
    for (int c = 0; c < 16; ++c) {
      float sc = s.ScoresF[row * NND + c0 + c];
      float al = (sc >= 0.f ? sc : SLOPE * sc) * bf2f(s.Adj[row * NND + c0 + c]);
      av[c] = al;
      m = fmaxf(m, al);
    }
    m = fmaxf(m, __shfl_xor(m, 1, 32));
    m = fmaxf(m, __shfl_xor(m, 2, 32));
    float ssum = 0.f;
#pragma unroll
    for (int c = 0; c < 16; ++c) { float e = __expf(av[c] - m); av[c] = e; ssum += e; }
    ssum += __shfl_xor(ssum, 1, 32);
    ssum += __shfl_xor(ssum, 2, 32);
    const float inv = 1.f / ssum;
#pragma unroll
    for (int c = 0; c < 16; ++c) s.Attn[row * NND + c0 + c] = f2bf(av[c] * inv);
  }
  __syncthreads();

  // -------- Phase E: hp = attn @ Whk (K = 64 = 2x32) --------
  f32x8 accH[4];
#pragma unroll
  for (int t = 0; t < 4; ++t) accH[t] = zero8();
#pragma unroll
  for (int ks = 0; ks < 2; ++ks) {
    bf16x16 a = load_frag_a(s.Attn, NND, mBase, ks * 32, l, hi);
#pragma unroll
    for (int t = 0; t < 4; ++t) {
      bf16x16 bb = load_frag_b(s.WhkT, NND, ks * 32, (nb0 + t) * 16, l, hi);
      accH[t] = wmma_bf16(a, bb, accH[t]);
    }
  }

  // -------- Phase F: gates (K padded to 64); gate hp; stash hp to LDS --------
  f32x8 accGd[4], accGg[4];
#pragma unroll
  for (int t = 0; t < 4; ++t) { accGd[t] = zero8(); accGg[t] = zero8(); }
#pragma unroll
  for (int ks = 0; ks < 2; ++ks) {
    bf16x16 a = load_frag_a(s.Op, OPDP, mBase, ks * 32, l, hi);
#pragma unroll
    for (int t = 0; t < 4; ++t) {
      bf16x16 bd = load_frag_b(s.WopDt, OPDP, ks * 32, (nb0 + t) * 16, l, hi);
      accGd[t] = wmma_bf16(a, bd, accGd[t]);
      bf16x16 bg = load_frag_b(s.WopGt, OPDP, ks * 32, (nb0 + t) * 16, l, hi);
      accGg[t] = wmma_bf16(a, bg, accGg[t]);
    }
  }
#pragma unroll
  for (int t = 0; t < 4; ++t) {
    const int o = (nb0 + t) * 16 + l;
#pragma unroll
    for (int r = 0; r < 8; ++r) {
      const int i = mBase + r + 8 * hi;
      const float gg  = 1.f / (1.f + __expf(-(accGg[t][r] + s.bopg[o])));
      const float hpv = gg * accH[t][r];
      accH[t][r] = hpv;
      s.HpF[i * FOUTN + o] = hpv;
      accGd[t][r] = 1.f / (1.f + __expf(-(accGd[t][r] + s.bopd[o])));  // sigmoid in place
    }
  }
  __syncthreads();

  // -------- Phase G: LayerNorm statistics per row (mean / rsqrt(var+eps)) --------
  {
    const int row = tid >> 2;
    const int c0  = (tid & 3) * 32;
    float sum = 0.f, sq = 0.f;
#pragma unroll
    for (int c = 0; c < 32; ++c) {
      float v = s.HpF[row * FOUTN + c0 + c];
      sum += v; sq += v * v;
    }
    sum += __shfl_xor(sum, 1, 32); sum += __shfl_xor(sum, 2, 32);
    sq  += __shfl_xor(sq, 1, 32);  sq  += __shfl_xor(sq, 2, 32);
    if ((tid & 3) == 0) {
      const float muv  = sum * (1.f / 128.f);
      const float varv = sq * (1.f / 128.f) - muv * muv;
      s.mu[row]   = muv;
      s.rstd[row] = rsqrtf(varv + EPSL);
    }
  }
  __syncthreads();

  // -------- Phase H: out = (gate_d*adjS + support + b + LN(hp)*gamma+beta)/2 ----
  float* outB = outG + (size_t)bId * NND * FOUTN;
#pragma unroll
  for (int t = 0; t < 4; ++t) {
    const int o = (nb0 + t) * 16 + l;
#pragma unroll
    for (int r = 0; r < 8; ++r) {
      const int i = mBase + r + 8 * hi;
      const float dense = accGd[t][r] * accA[t][r] + accS[t][r] + s.bvec[o];
      const float hpn   = (accH[t][r] - s.mu[i]) * s.rstd[i] * s.gam[o] + s.bet[o];
      outB[i * FOUTN + o] = 0.5f * (dense + hpn);
    }
  }
}

extern "C" void kernel_launch(void* const* d_in, const int* in_sizes, int n_in,
                              void* d_out, int out_size, void* d_ws, size_t ws_size,
                              hipStream_t stream) {
  (void)n_in; (void)out_size; (void)d_ws; (void)ws_size;
  const float* h     = (const float*)d_in[0];
  const float* adj   = (const float*)d_in[1];
  const float* op    = (const float*)d_in[2];
  const float* W     = (const float*)d_in[3];
  const float* bvec  = (const float*)d_in[4];
  const float* WopD  = (const float*)d_in[5];
  const float* bopD  = (const float*)d_in[6];
  const float* Wk    = (const float*)d_in[7];
  const float* aw    = (const float*)d_in[8];
  const float* WopG  = (const float*)d_in[9];
  const float* bopG  = (const float*)d_in[10];
  const float* gam   = (const float*)d_in[11];
  const float* bet   = (const float*)d_in[12];
  float* out = (float*)d_out;

  const int B = in_sizes[0] / (NND * FINN);          // 2048
  const size_t shmem = sizeof(Smem);                 // ~232 KB, within 320 KB/WGP
  (void)hipFuncSetAttribute(reinterpret_cast<const void*>(&gin_fused_kernel),
                            hipFuncAttributeMaxDynamicSharedMemorySize, (int)shmem);
  gin_fused_kernel<<<B, 256, shmem, stream>>>(h, adj, op, W, bvec, WopD, bopD,
                                              Wk, aw, WopG, bopG, gam, bet, out);
}
